// IONet_74698071212299
// MI455X (gfx1250) — compile-verified, matrix operand order checked by reading, not run on hardware
//
#include <hip/hip_runtime.h>
#include <math.h>

// ---------------------------------------------------------------------------
// Fused IONet forward for MI455X (gfx1250, wave32, WMMA).
// D = A(16x32 f16) x B(32x16 f16) + C(16x16 f32) via v_wmma_f32_16x16x32_f16.
// Memory-bound (~647MB streamed @ 23.3TB/s => ~28us floor); f16 matrix path
// with f32 accumulation keeps compute well under that floor.
// Fixes: hw/branch-free tanh (device-gated), no exec-masked conditional loads
// (clamp+select), b128 LDS fragment gathers (ISA layout is contiguous per
// 8-half group), b128 input staging.
// ---------------------------------------------------------------------------

typedef __attribute__((ext_vector_type(16))) _Float16 v16h;
typedef __attribute__((ext_vector_type(8)))  float    v8f;

#define WAVES   2
#define THREADS (WAVES * 32)
#define WIDTH   20
#define K0      101
#define K0S     128     // layer-0 K storage (zero padded), branches
#define KHS     32      // hidden-layer K storage (20 padded to 32)

// f16 W^T image layout (halves):
//   branch net i (i=0..3): base i*5120 : [20][128] L1, then 4 x [20][32]
//   trunk (net 4):         base 20480  : 5 x [20][32]
#define BR_NET_H   5120
#define TR_BASE_H  (4 * BR_NET_H)          // 20480
#define SW_HALVES  (TR_BASE_H + 5 * 640)   // 23680 halves = 47360 bytes
#define SB_FLOATS  (25 * 32)               // padded biases, 3200 bytes

struct NetPtrs { const float* W[25]; const float* b[25]; };

union AF { v16h v; unsigned u[8]; uint4 q[2]; };
union HPK { _Float16 f[2]; unsigned w; };
union QPK { _Float16 f[8]; uint4 q; };

__device__ __forceinline__ int wOff(int net, int l) {
  return (net < 4) ? (net * BR_NET_H + ((l == 0) ? 0 : 2560 + (l - 1) * 640))
                   : (TR_BASE_H + l * 640);
}

// Branch-free tanh. CDNA5 has a TANH transcendental; fall back to
// 1 - 2*rcp(1+exp2(2*log2e*x)) (saturates correctly at +-1, no branches).
// Host pass only needs this to parse; gate amdgcn builtins on device compile.
__device__ __forceinline__ float fast_tanh(float x) {
#if defined(__HIP_DEVICE_COMPILE__)
#if __has_builtin(__builtin_amdgcn_tanhf)
  return __builtin_amdgcn_tanhf(x);
#elif __has_builtin(__builtin_amdgcn_exp2f) && __has_builtin(__builtin_amdgcn_rcpf)
  const float t = __builtin_amdgcn_exp2f(x * 2.885390081777927f);
  return 1.0f - 2.0f * __builtin_amdgcn_rcpf(1.0f + t);
#else
  const float t = exp2f(x * 2.885390081777927f);
  return 1.0f - 2.0f / (1.0f + t);
#endif
#else
  return tanhf(x);
#endif
}

// ISA 16-bit A-matrix 16x32 layout: lane L holds M=L&15, half h=L>>4;
// vector elements 0..7 are halves k = kb+h*8 .. +7, elements 8..15 are
// k = kb+16+h*8 .. +7  => two contiguous 16B groups => 2x ds_load_b128.
__device__ __forceinline__ AF gatherA(const _Float16* base, int pitch, int m,
                                      int h, int kb) {
  const _Float16* p = base + m * pitch + kb + h * 8;
  AF a;
  a.q[0] = *(const uint4*)p;
  a.q[1] = *(const uint4*)(p + 16);
  return a;
}

// B (KxN) from W^T stored [n][Kstore] (mirrored layout, lane -> N).
// Clamp row + AND-mask to zero for padded columns: no exec-masked loads.
__device__ __forceinline__ AF gatherB(const _Float16* w, int pitch, int ncol,
                                      int h, int kb) {
  const int nc = (ncol < WIDTH) ? ncol : (WIDTH - 1);
  const unsigned msk = (ncol < WIDTH) ? 0xFFFFFFFFu : 0u;
  const _Float16* p = w + nc * pitch + kb + h * 8;
  AF b;
  b.q[0] = *(const uint4*)p;
  b.q[1] = *(const uint4*)(p + 16);
#pragma unroll
  for (int j = 0; j < 8; ++j) b.u[j] &= msk;
  return b;
}

__device__ __forceinline__ v8f wmma16(const AF& a, const AF& b, v8f c) {
  return __builtin_amdgcn_wmma_f32_16x16x32_f16(false, a.v, false, b.v,
                                                (short)0, c, false, false);
}

// Convert all weights to padded f16 W^T + padded f32 biases (clamped loads).
__device__ void stage_weights(const NetPtrs& P, _Float16* dW, float* dB,
                              int tid, int nthr) {
  for (int net = 0; net < 5; ++net) {
    for (int l = 0; l < 5; ++l) {
      const int K  = (l == 0) ? ((net < 4) ? K0 : 2) : WIDTH;
      const int KS = (l == 0 && net < 4) ? K0S : KHS;
      _Float16* dst = dW + wOff(net, l);
      const float* Wg = P.W[net * 5 + l];
      for (int idx = tid; idx < WIDTH * KS; idx += nthr) {
        const int nn = idx / KS, k = idx - nn * KS;
        const int kc = (k < K) ? k : (K - 1);
        const float wv = Wg[kc * WIDTH + nn];
        dst[idx] = (k < K) ? (_Float16)wv : (_Float16)0.f;
      }
      const float* bg = P.b[net * 5 + l];
      for (int idx = tid; idx < 32; idx += nthr) {
        const float bv = bg[(idx < WIDTH) ? idx : 0];
        dB[(net * 5 + l) * 32 + idx] = (idx < WIDTH) ? bv : 0.f;
      }
    }
  }
}

__global__ void prep_weights_kernel(NetPtrs P, _Float16* ws) {
  stage_weights(P, ws, (float*)(ws + SW_HALVES), threadIdx.x, blockDim.x);
}

// Layers 1..4 given layer-0 pre-activation fragments (bias not yet added).
__device__ __forceinline__ void hidden_chain(v8f c0, v8f c1,
                                             const _Float16* wb1,  // layers 1..4
                                             const float* bb,      // net bias base
                                             _Float16* act, int lane,
                                             v8f& o0, v8f& o1) {
  const int n0 = lane & 15, h = lane >> 4;
  const v8f Z = {0.f, 0.f, 0.f, 0.f, 0.f, 0.f, 0.f, 0.f};
#pragma unroll
  for (int l = 0; l < 4; ++l) {
    const float b0 = bb[l * 32 + n0];
    const float b1 = bb[l * 32 + 16 + n0];
    asm volatile("s_wait_dscnt 0x0" ::: "memory");  // prior act reads retired
#pragma unroll
    for (int r = 0; r < 8; ++r) {
      act[(r + 8 * h) * KHS + n0]      = (_Float16)fast_tanh(c0[r] + b0);
      act[(r + 8 * h) * KHS + 16 + n0] = (_Float16)fast_tanh(c1[r] + b1);
    }
    asm volatile("s_wait_dscnt 0x0" ::: "memory");  // stores visible wave-wide
    const AF a = gatherA(act, KHS, n0, h, 0);
    const _Float16* w = wb1 + l * (WIDTH * KHS);
    c0 = wmma16(a, gatherB(w, KHS, n0, h, 0), Z);
    c1 = wmma16(a, gatherB(w, KHS, 16 + n0, h, 0), Z);
  }
  const float b0 = bb[4 * 32 + n0];
  const float b1 = bb[4 * 32 + 16 + n0];
#pragma unroll
  for (int r = 0; r < 8; ++r) { o0[r] = c0[r] + b0; o1[r] = c1[r] + b1; }
}

template <bool FROM_WS>
__global__ __launch_bounds__(THREADS)
void ionet_kernel(NetPtrs P, const _Float16* __restrict__ wsW,
                  const float* __restrict__ bu, const float* __restrict__ bd,
                  const float* __restrict__ al, const float* __restrict__ ar,
                  const float* __restrict__ xin, const float* __restrict__ bias1,
                  float* __restrict__ out, int N) {
  __shared__ __align__(16) _Float16 sW[SW_HALVES];
  __shared__ __align__(16) float    sB[SB_FLOATS];
  __shared__ __align__(16) _Float16 sIn[WAVES][16 * K0S];
  __shared__ __align__(16) _Float16 sAct[WAVES][16 * KHS];

  const int tid = threadIdx.x;

  if (FROM_WS) {  // fast path: block-copy prebuilt f16 image (L2 resident)
    const uint4* srcW = (const uint4*)wsW;
    uint4* dW = (uint4*)sW;
    for (int idx = tid; idx < SW_HALVES / 8; idx += THREADS) dW[idx] = srcW[idx];
    const uint4* srcB = (const uint4*)(wsW + SW_HALVES);
    uint4* dB = (uint4*)sB;
    for (int idx = tid; idx < SB_FLOATS / 4; idx += THREADS) dB[idx] = srcB[idx];
  } else {
    stage_weights(P, sW, sB, tid, THREADS);
  }
  __syncthreads();

  const int wave = tid >> 5, lane = tid & 31;
  const int n0 = lane & 15, h = lane >> 4;
  const int ntiles = (N + 15) >> 4;
  const int gwaves = gridDim.x * WAVES;
  const float bias1v = bias1[0];
  const v8f Z = {0.f, 0.f, 0.f, 0.f, 0.f, 0.f, 0.f, 0.f};
  _Float16* si  = sIn[wave];
  _Float16* act = sAct[wave];

  // Zero the constant pad region of the input tile once (halves 104..127).
  for (int idx = lane; idx < 16 * 3; idx += 32) {
    const int m = idx / 3, c = 13 + (idx - m * 3);
    *(uint4*)(si + m * K0S + c * 8) = make_uint4(0u, 0u, 0u, 0u);
  }

  for (int tile = blockIdx.x * WAVES + wave; tile < ntiles; tile += gwaves) {
    const long row0 = (long)tile * 16;

    // ---- trunk: K=2 lives entirely in A element-pair 0 of lanes 0..15 ----
    v8f acc0, acc1;
    {
      AF a;
      a.q[0] = make_uint4(0u, 0u, 0u, 0u);
      a.q[1] = make_uint4(0u, 0u, 0u, 0u);
      if (h == 0) {
        long r = row0 + n0; if (r >= N) r = N - 1;
        const float* xr = xin + r * 2;
        HPK pk; pk.f[0] = (_Float16)xr[0]; pk.f[1] = (_Float16)xr[1];
        a.u[0] = pk.w;
      }
      const _Float16* w0 = sW + wOff(4, 0);
      v8f c0 = wmma16(a, gatherB(w0, KHS, n0, h, 0), Z);
      v8f c1 = wmma16(a, gatherB(w0, KHS, 16 + n0, h, 0), Z);
      hidden_chain(c0, c1, sW + wOff(4, 1), sB + 4 * 160, act, lane, acc0, acc1);
    }

    // ---- four branch nets, multiplied into the accumulator fragments ----
#pragma unroll 1
    for (int bnet = 0; bnet < 4; ++bnet) {
      const float* src = (bnet == 0) ? bu : (bnet == 1) ? bd
                       : (bnet == 2) ? al : ar;
      asm volatile("s_wait_dscnt 0x0" ::: "memory");
      // Stage 16x101 f32 -> f16, b128 chunks of 8 halves.
      // Chunks 0..11 (k=0..95): fully in-bounds, no selects.
      for (int idx = lane; idx < 16 * 12; idx += 32) {
        const int m = idx / 12, c = idx - m * 12;
        long r = row0 + m; if (r >= N) r = N - 1;
        const float* p = src + (size_t)r * K0 + c * 8;
        QPK pk;
#pragma unroll
        for (int j = 0; j < 8; ++j) pk.f[j] = (_Float16)p[j];
        *(uint4*)(si + m * K0S + c * 8) = pk.q;
      }
      // Chunk 12 (k=96..103): 5 valid halves, compile-time zero padding.
      if (lane < 16) {
        const int m = lane;
        long r = row0 + m; if (r >= N) r = N - 1;
        const float* p = src + (size_t)r * K0;
        QPK pk;
#pragma unroll
        for (int j = 0; j < 8; ++j)
          pk.f[j] = (96 + j < K0) ? (_Float16)p[96 + j] : (_Float16)0.f;
        *(uint4*)(si + m * K0S + 96) = pk.q;
      }
      asm volatile("s_wait_dscnt 0x0" ::: "memory");

      // Layer 0: 4 k-steps x 2 n-tiles; interleave the two accum chains.
      AF a[4];
#pragma unroll
      for (int kt = 0; kt < 4; ++kt) a[kt] = gatherA(si, K0S, n0, h, kt * 32);
      const _Float16* w0 = sW + wOff(bnet, 0);
      v8f c0 = Z, c1 = Z;
#pragma unroll
      for (int kt = 0; kt < 4; ++kt) {
        const AF b0 = gatherB(w0, K0S, n0, h, kt * 32);
        const AF b1 = gatherB(w0, K0S, 16 + n0, h, kt * 32);
        c0 = wmma16(a[kt], b0, c0);
        c1 = wmma16(a[kt], b1, c1);
      }

      v8f o0, o1;
      hidden_chain(c0, c1, sW + wOff(bnet, 1), sB + bnet * 160, act, lane, o0, o1);
      acc0 = acc0 * o0;
      acc1 = acc1 * o1;
    }

    // ---- reduce over width (n=0..19) and store [N,1] ----
#pragma unroll
    for (int r = 0; r < 8; ++r) {
      float t = acc0[r] + ((n0 < 4) ? acc1[r] : 0.f);
      t += __shfl_xor(t, 1, 16);
      t += __shfl_xor(t, 2, 16);
      t += __shfl_xor(t, 4, 16);
      t += __shfl_xor(t, 8, 16);
      const long orow = row0 + r + 8 * h;   // lanes0-15: rows r; lanes16-31: r+8
      if (n0 == 0 && orow < N) out[orow] = t + bias1v;
    }
  }
}

extern "C" void kernel_launch(void* const* d_in, const int* in_sizes, int n_in,
                              void* d_out, int out_size, void* d_ws, size_t ws_size,
                              hipStream_t stream) {
  (void)n_in; (void)out_size;
  NetPtrs P;
  // dict order: 4 boundary arrays, x, then params1..4 (5x(W,b) each), trunk, bias1
  for (int net = 0; net < 5; ++net)
    for (int l = 0; l < 5; ++l) {
      P.W[net * 5 + l] = (const float*)d_in[5 + net * 10 + 2 * l];
      P.b[net * 5 + l] = (const float*)d_in[5 + net * 10 + 2 * l + 1];
    }
  const float* bu    = (const float*)d_in[0];
  const float* bd    = (const float*)d_in[1];
  const float* al    = (const float*)d_in[2];
  const float* ar    = (const float*)d_in[3];
  const float* xin   = (const float*)d_in[4];
  const float* bias1 = (const float*)d_in[55];
  float* out = (float*)d_out;

  const int N = in_sizes[0] / K0;
  const int ntiles = (N + 15) / 16;
  int blocks = (ntiles + WAVES - 1) / WAVES;
  if (blocks > 2048) blocks = 2048;

  const size_t wsNeed = (size_t)SW_HALVES * 2 + (size_t)SB_FLOATS * 4;  // 50560 B
  if (ws_size >= wsNeed) {
    hipLaunchKernelGGL(prep_weights_kernel, dim3(1), dim3(256), 0, stream,
                       P, (_Float16*)d_ws);
    hipLaunchKernelGGL((ionet_kernel<true>), dim3(blocks), dim3(THREADS), 0, stream,
                       P, (const _Float16*)d_ws, bu, bd, al, ar, xin, bias1, out, N);
  } else {
    hipLaunchKernelGGL((ionet_kernel<false>), dim3(blocks), dim3(THREADS), 0, stream,
                       P, (const _Float16*)nullptr, bu, bd, al, ar, xin, bias1, out, N);
  }
}
